// Qwen3_5Attention_85529978732976
// MI455X (gfx1250) — matrix operand order repaired and checked by reading, not compiled, not run on hardware
//
#include <hip/hip_runtime.h>
#include <hip/hip_bf16.h>

// ---------------------------------------------------------------------------
// Qwen3.5-style gated attention block for MI455X (gfx1250, wave32, WMMA).
// All matmuls use V_WMMA_F32_16X16X32_BF16. All LDS staging runs through the
// CDNA5 async data mover (GLOBAL_LOAD_ASYNC_TO_LDS_B128, ASYNCcnt) with a
// double-buffered pipeline: the next tile's async copies are in flight while
// the current tile feeds the matrix pipes.
// ---------------------------------------------------------------------------

typedef __bf16 bf16;
typedef __attribute__((ext_vector_type(16))) __bf16 v16bf;
typedef __attribute__((ext_vector_type(8)))  float  v8f;

#define HIDDEN   2048
#define NH       16
#define NKV      8
#define HD       128
#define QKV_N    6144      // 4096 (q+gate) + 1024 (k) + 1024 (v)
#define SEQQ     2048
#define BATCH    2
#define TOK      (BATCH * SEQQ)
#define EPS_RMS  1e-6f
#define THETA    1.0e7f

// ---------------------------------------------------------------------------
// helpers
// ---------------------------------------------------------------------------
__device__ __forceinline__ void storeOut(float* p, float v) { *p = v; }
__device__ __forceinline__ void storeOut(bf16*  p, float v) { *p = (bf16)v; }

// CDNA5 async copy: each active lane copies 16B global -> 16B LDS, tracked by
// ASYNCcnt. LDS address = low 32 bits of the generic pointer (flat LDS
// aperture truncates to the wave-relative offset, ISA 10.2).
__device__ __forceinline__ void async_b128(void* lds, const void* gptr) {
  uint32_t l = (uint32_t)(uintptr_t)lds;
  asm volatile("global_load_async_to_lds_b128 %0, %1, off"
               :: "v"(l), "v"(gptr) : "memory");
}
// Async loads complete in order: with 4 ops/tile, waiting ASYNCcnt<=4 after
// issuing the next tile guarantees the current tile has landed.
__device__ __forceinline__ void wait_async0()   { asm volatile("s_wait_asynccnt 0" ::: "memory"); }
__device__ __forceinline__ void wait_async_le4(){ asm volatile("s_wait_asynccnt 4" ::: "memory"); }

union FragPA { uint4 q[2]; v16bf v; };

// ---------------------------------------------------------------------------
// fp32 -> bf16 conversion
// ---------------------------------------------------------------------------
__global__ void k_cvt_bf16(const float* __restrict__ src, bf16* __restrict__ dst, int n) {
  int i = blockIdx.x * blockDim.x + threadIdx.x;
  if (i < n) dst[i] = (bf16)src[i];
}

// Concatenate Wq | Wk | Wv into one [HIDDEN][6144] bf16 matrix so QKV is one GEMM.
__global__ void k_build_wcat(const float* __restrict__ Wq, const float* __restrict__ Wk,
                             const float* __restrict__ Wv, bf16* __restrict__ Wcat) {
  int i = blockIdx.x * blockDim.x + threadIdx.x;
  if (i >= HIDDEN * QKV_N) return;
  int row = i / QKV_N, col = i - row * QKV_N;
  float v;
  if (col < 4096)       v = Wq[row * 4096 + col];
  else if (col < 5120)  v = Wk[row * 1024 + (col - 4096)];
  else                  v = Wv[row * 1024 + (col - 5120)];
  Wcat[i] = (bf16)v;
}

// ---------------------------------------------------------------------------
// Tiled bf16 GEMM: C[M][N] = A[M][K] * B[K][N], f32 accum via WMMA.
// Block tile 128x128, 8 waves (4x2), each wave 32x64 = 2x4 WMMA tiles,
// K-step 32. Double-buffered async staging.
// ---------------------------------------------------------------------------
template <typename OutT>
__global__ __launch_bounds__(256) void k_gemm(const bf16* __restrict__ A,
                                              const bf16* __restrict__ B,
                                              OutT* __restrict__ C,
                                              int M, int N, int K) {
  __shared__ bf16 As[2][128][40];   // pads keep 16B alignment, avoid bank conflicts
  __shared__ bf16 Bs[2][32][136];

  const int tid  = threadIdx.x;
  const int lane = tid & 31;
  const int wid  = tid >> 5;
  const int hi   = lane >> 4;       // half-wave select
  const int l16  = lane & 15;
  const int mBase = blockIdx.y * 128;
  const int nBase = blockIdx.x * 128;
  const int wm = (wid >> 1) * 32;   // wave row offset (0/32/64/96)
  const int wn = (wid & 1) * 64;    // wave col offset (0/64)

  const int ar = tid >> 1, ah = (tid & 1) * 16;   // A loaders: 2 threads/row, 16 elems
  const int br = tid >> 3, bc = (tid & 7) * 16;   // B loaders: 8 threads/row, 16 elems

  auto stage = [&](int k0, int p) {   // 4 async b128 ops per thread
    const bf16* aSrc = A + (size_t)(mBase + ar) * K + k0 + ah;
    const bf16* bSrc = B + (size_t)(k0 + br) * N + nBase + bc;
    async_b128(&As[p][ar][ah],     aSrc);
    async_b128(&As[p][ar][ah + 8], aSrc + 8);
    async_b128(&Bs[p][br][bc],     bSrc);
    async_b128(&Bs[p][br][bc + 8], bSrc + 8);
  };

  v8f acc[2][4] = {};

  stage(0, 0);
  int p = 0;
  for (int k0 = 0; k0 < K; k0 += 32, p ^= 1) {
    const bool more = (k0 + 32 < K);
    if (more) stage(k0 + 32, p ^ 1);      // overlap next tile with this compute
    if (more) wait_async_le4(); else wait_async0();
    __syncthreads();

    // A fragment: 16x32, lane holds row M=l16 (CDNA5 ISA 7.12.2 layout)
    v16bf af[2], bfr[4];
#pragma unroll
    for (int mi = 0; mi < 2; ++mi) {
      int m = wm + mi * 16 + l16;
      FragPA fa;
      fa.q[0] = *(const uint4*)&As[p][m][hi * 8];
      fa.q[1] = *(const uint4*)&As[p][m][16 + hi * 8];
      af[mi] = fa.v;
    }
    // B fragment: 32x16, lane holds col N=l16; K = hi*8..hi*8+7, 16+hi*8..16+hi*8+7
#pragma unroll
    for (int ni = 0; ni < 4; ++ni) {
      int n = wn + ni * 16 + l16;
#pragma unroll
      for (int i = 0; i < 16; ++i) bfr[ni][i] = Bs[p][(i & 7) + ((i >> 3) << 4) + hi * 8][n];
    }
#pragma unroll
    for (int mi = 0; mi < 2; ++mi)
#pragma unroll
      for (int ni = 0; ni < 4; ++ni)
        acc[mi][ni] = __builtin_amdgcn_wmma_f32_16x16x32_bf16(
            false, af[mi], false, bfr[ni], (short)0, acc[mi][ni], false, false);
    __syncthreads();   // WAR: protect the buffer the next iteration restages
  }

  // C/D layout: element r of v8f is row M = r + 8*hi, col N = l16
#pragma unroll
  for (int mi = 0; mi < 2; ++mi)
#pragma unroll
    for (int ni = 0; ni < 4; ++ni)
#pragma unroll
      for (int r = 0; r < 8; ++r) {
        int row = mBase + wm + mi * 16 + r + hi * 8;
        int col = nBase + wn + ni * 16 + l16;
        storeOut(&C[(size_t)row * N + col], acc[mi][ni][r]);
      }
}

// ---------------------------------------------------------------------------
// Per-token epilogue: RMSnorm(q, k) with (1+w) weight, then RoPE on dims 0..31.
// ---------------------------------------------------------------------------
__global__ __launch_bounds__(256) void k_norm_rope(const bf16* __restrict__ QKV,
                                                   const float* __restrict__ qw,
                                                   const float* __restrict__ kw,
                                                   const int* __restrict__ positions,
                                                   bf16* __restrict__ Qr,
                                                   bf16* __restrict__ Kr) {
  __shared__ float sq[NH][HD];
  __shared__ float sk[NKV][HD];
  const int t   = blockIdx.x;
  const int tid = threadIdx.x;
  const float pos = (float)positions[t];
  const bf16* row = QKV + (size_t)t * QKV_N;

  for (int i = tid; i < NH * HD; i += 256) {
    int h = i >> 7, d = i & 127;
    sq[h][d] = (float)row[h * 256 + d];        // q = first 128 of each 256-block
  }
  for (int i = tid; i < NKV * HD; i += 256)
    sk[i >> 7][i & 127] = (float)row[4096 + i];
  __syncthreads();

  {  // Q norm: 16 heads x 16 threads (half-wave groups), 8 dims each
    int h = tid >> 4, g = tid & 15;
    float ss = 0.f;
#pragma unroll
    for (int j = 0; j < 8; ++j) { float v = sq[h][g * 8 + j]; ss += v * v; }
    ss += __shfl_xor(ss, 1, 32); ss += __shfl_xor(ss, 2, 32);
    ss += __shfl_xor(ss, 4, 32); ss += __shfl_xor(ss, 8, 32);
    float r = rsqrtf(ss * (1.0f / HD) + EPS_RMS);
#pragma unroll
    for (int j = 0; j < 8; ++j) {
      int d = g * 8 + j;
      sq[h][d] = sq[h][d] * r * (1.0f + qw[d]);
    }
  }
  {  // K norm: 8 heads x 32 threads (full-wave groups), 4 dims each
    int h = tid >> 5, g = tid & 31;
    float ss = 0.f;
#pragma unroll
    for (int j = 0; j < 4; ++j) { float v = sk[h][g * 4 + j]; ss += v * v; }
    ss += __shfl_xor(ss, 1, 32);  ss += __shfl_xor(ss, 2, 32);
    ss += __shfl_xor(ss, 4, 32);  ss += __shfl_xor(ss, 8, 32);
    ss += __shfl_xor(ss, 16, 32);
    float r = rsqrtf(ss * (1.0f / HD) + EPS_RMS);
#pragma unroll
    for (int j = 0; j < 4; ++j) {
      int d = g * 4 + j;
      sk[h][d] = sk[h][d] * r * (1.0f + kw[d]);
    }
  }
  __syncthreads();

  for (int i = tid; i < NH * HD; i += 256) {
    int h = i >> 7, d = i & 127;
    float o;
    if (d < 16) {
      float a = pos * __powf(THETA, -(float)d / 16.0f);
      o = sq[h][d] * __cosf(a) - sq[h][d + 16] * __sinf(a);
    } else if (d < 32) {
      int d0 = d - 16;
      float a = pos * __powf(THETA, -(float)d0 / 16.0f);
      o = sq[h][d0] * __sinf(a) + sq[h][d] * __cosf(a);
    } else o = sq[h][d];
    Qr[(size_t)t * (NH * HD) + i] = (bf16)o;
  }
  for (int i = tid; i < NKV * HD; i += 256) {
    int h = i >> 7, d = i & 127;
    float o;
    if (d < 16) {
      float a = pos * __powf(THETA, -(float)d / 16.0f);
      o = sk[h][d] * __cosf(a) - sk[h][d + 16] * __sinf(a);
    } else if (d < 32) {
      int d0 = d - 16;
      float a = pos * __powf(THETA, -(float)d0 / 16.0f);
      o = sk[h][d0] * __sinf(a) + sk[h][d] * __cosf(a);
    } else o = sk[h][d];
    Kr[(size_t)t * (NKV * HD) + i] = (bf16)o;
  }
}

// ---------------------------------------------------------------------------
// Flash attention with online softmax. One block = one (batch, head, 128 q
// rows); all 8 waves share block-staged K/V tiles (double-buffered async
// copies). Per 32-key step and wave: S(16x32)=QK^T (8 wmma), online softmax,
// P->LDS relayout, O(16x128)+=PV (8 wmma). Epilogue fuses 1/l + sigmoid(gate).
// ---------------------------------------------------------------------------
__global__ __launch_bounds__(256) void k_attn(const bf16* __restrict__ Qr,
                                              const bf16* __restrict__ Kr,
                                              const bf16* __restrict__ QKV,   // V + gate live here
                                              const int* __restrict__ amask,
                                              bf16* __restrict__ AttnG) {
  __shared__ bf16 Ks[2][32][136];    // 32 keys x 128 dims (+pad), double-buffered
  __shared__ bf16 Vs[2][32][136];
  __shared__ bf16 Pbuf[8][16][32];   // wave-private P staging (C-layout -> A-layout)

  const int tid = threadIdx.x, lane = tid & 31, wid = tid >> 5;
  const int hi = lane >> 4, l16 = lane & 15;
  const int h   = blockIdx.y;        // q head
  const int b   = blockIdx.z;
  const int hkv = h >> 1;            // 16 q heads -> 8 kv heads (g = 2)
  const int qb  = blockIdx.x * 128;  // block q row base (per batch)
  const int q0  = qb + wid * 16;     // wave q row base
  const size_t tokBase = (size_t)b * SEQQ;

  // staging roles: 32 rows x 8 threads, 16 dims per thread
  const int sr = tid >> 3, sd = (tid & 7) * 16;

  auto stageKV = [&](int kk, int pp) {   // 4 async b128 ops per thread
    const bf16* kSrc = Kr  + (tokBase + kk + sr) * (size_t)(NKV * HD) + hkv * HD + sd;
    const bf16* vSrc = QKV + (tokBase + kk + sr) * (size_t)QKV_N + 5120 + hkv * HD + sd;
    async_b128(&Ks[pp][sr][sd],     kSrc);
    async_b128(&Ks[pp][sr][sd + 8], kSrc + 8);
    async_b128(&Vs[pp][sr][sd],     vSrc);
    async_b128(&Vs[pp][sr][sd + 8], vSrc + 8);
  };

  // Q A-fragments: 4 K-chunks of 32 along head_dim
  v16bf aq[4];
  {
    const bf16* qp = Qr + (tokBase + q0 + l16) * (size_t)(NH * HD) + h * HD;
#pragma unroll
    for (int c = 0; c < 4; ++c)
#pragma unroll
      for (int i = 0; i < 8; ++i) {
        aq[c][i]     = qp[c * 32 + hi * 16 + i];
        aq[c][8 + i] = qp[c * 32 + hi * 16 + 8 + i];
      }
  }

  float mrow[8], lrow[8];
  v8f acc[8] = {};                   // 16 x 128 output accumulator
#pragma unroll
  for (int r = 0; r < 8; ++r) { mrow[r] = -1e30f; lrow[r] = 0.f; }

  const float scale = 0.08838834764831845f;  // 1/sqrt(128)
  const int kmaxw = q0 + 15;                 // causal bound for this wave
  const int kmaxb = qb + 127;                // causal bound for the block

  stageKV(0, 0);
  int p = 0;
  for (int kk = 0; kk <= kmaxb; kk += 32, p ^= 1) {
    const bool more = (kk + 32 <= kmaxb);
    if (more) stageKV(kk + 32, p ^ 1);       // overlap next tile with compute
    if (more) wait_async_le4(); else wait_async0();
    __syncthreads();

    if (kk <= kmaxw) {   // wave-uniform branch: EXEC stays all-ones for WMMA
      // ---- S = Q K^T for keys [kk, kk+32), two 16-wide N tiles ----
      v8f s0 = {}, s1 = {};
#pragma unroll
      for (int c = 0; c < 4; ++c) {
        v16bf bk0, bk1;   // B frag: lane = key col, 16 contiguous d per lane
        FragPA f0, f1;
        f0.q[0] = *(const uint4*)&Ks[p][l16][c * 32 + hi * 16];
        f0.q[1] = *(const uint4*)&Ks[p][l16][c * 32 + hi * 16 + 8];
        f1.q[0] = *(const uint4*)&Ks[p][16 + l16][c * 32 + hi * 16];
        f1.q[1] = *(const uint4*)&Ks[p][16 + l16][c * 32 + hi * 16 + 8];
        bk0 = f0.v; bk1 = f1.v;
        s0 = __builtin_amdgcn_wmma_f32_16x16x32_bf16(false, aq[c], false, bk0, (short)0, s0, false, false);
        s1 = __builtin_amdgcn_wmma_f32_16x16x32_bf16(false, aq[c], false, bk1, (short)0, s1, false, false);
      }

      // ---- masked online softmax update ----
      const int key0 = kk + l16, key1 = kk + 16 + l16;
      const float kv0 = (amask[b * SEQQ + key0] != 0) ? 0.f : -1e30f;
      const float kv1 = (amask[b * SEQQ + key1] != 0) ? 0.f : -1e30f;
#pragma unroll
      for (int r = 0; r < 8; ++r) {
        int qrow = q0 + r + hi * 8;
        float v0 = (key0 <= qrow) ? (s0[r] * scale + kv0) : -1e30f;
        float v1 = (key1 <= qrow) ? (s1[r] * scale + kv1) : -1e30f;
        float mt = fmaxf(v0, v1);
        mt = fmaxf(mt, __shfl_xor(mt, 1, 32)); mt = fmaxf(mt, __shfl_xor(mt, 2, 32));
        mt = fmaxf(mt, __shfl_xor(mt, 4, 32)); mt = fmaxf(mt, __shfl_xor(mt, 8, 32));
        float mnew = fmaxf(mrow[r], mt);
        float p0 = __expf(v0 - mnew), p1 = __expf(v1 - mnew);
        float sum = p0 + p1;
        sum += __shfl_xor(sum, 1, 32); sum += __shfl_xor(sum, 2, 32);
        sum += __shfl_xor(sum, 4, 32); sum += __shfl_xor(sum, 8, 32);
        float alpha = __expf(mrow[r] - mnew);
        lrow[r] = lrow[r] * alpha + sum;
        mrow[r] = mnew;
#pragma unroll
        for (int nd = 0; nd < 8; ++nd) acc[nd][r] *= alpha;
        Pbuf[wid][r + hi * 8][l16]      = (bf16)p0;   // C-layout -> row-major tile
        Pbuf[wid][r + hi * 8][16 + l16] = (bf16)p1;
      }

      // ---- re-read P as A-fragment (16 rows x 32 keys), vectorized ----
      FragPA pa;
      pa.q[0] = *(const uint4*)&Pbuf[wid][l16][hi * 8];
      pa.q[1] = *(const uint4*)&Pbuf[wid][l16][16 + hi * 8];

      // ---- O += P @ V from LDS ----
#pragma unroll
      for (int nd = 0; nd < 8; ++nd) {
        v16bf bv;   // B frag: lane = d col, 16 key rows per lane
#pragma unroll
        for (int i = 0; i < 16; ++i) bv[i] = Vs[p][(i & 7) + ((i >> 3) << 4) + hi * 8][nd * 16 + l16];
        acc[nd] = __builtin_amdgcn_wmma_f32_16x16x32_bf16(false, pa.v, false, bv, (short)0, acc[nd], false, false);
      }
    }
    __syncthreads();   // WAR: protect tiles before the next restage
  }

  // ---- epilogue: normalize by l, apply sigmoid gate, store bf16 ----
#pragma unroll
  for (int r = 0; r < 8; ++r) {
    float inv = 1.0f / lrow[r];
    size_t tok = tokBase + q0 + r + hi * 8;
    const bf16* gp = QKV + tok * (size_t)QKV_N + h * 256 + 128;  // gate = second 128
    bf16* op = AttnG + tok * (size_t)(NH * HD) + h * HD;
#pragma unroll
    for (int nd = 0; nd < 8; ++nd) {
      int d = nd * 16 + l16;
      float g  = (float)gp[d];
      float sg = 1.0f / (1.0f + __expf(-g));
      op[d] = (bf16)(acc[nd][r] * inv * sg);
    }
  }
}

// ---------------------------------------------------------------------------
// host-side launcher
// ---------------------------------------------------------------------------
extern "C" void kernel_launch(void* const* d_in, const int* in_sizes, int n_in,
                              void* d_out, int out_size, void* d_ws, size_t ws_size,
                              hipStream_t stream) {
  (void)in_sizes; (void)n_in; (void)out_size; (void)ws_size;
  const float* x         = (const float*)d_in[0];
  const int*   amask     = (const int*)  d_in[1];
  const int*   positions = (const int*)  d_in[2];
  const float* Wq        = (const float*)d_in[3];
  const float* Wk        = (const float*)d_in[4];
  const float* Wv        = (const float*)d_in[5];
  const float* Wo        = (const float*)d_in[6];
  const float* qw        = (const float*)d_in[7];
  const float* kw        = (const float*)d_in[8];
  float* out = (float*)d_out;

  // workspace carve-up (all offsets are multiples of 256B)
  char* p = (char*)d_ws;
  bf16* Xb    = (bf16*)p; p += (size_t)TOK * HIDDEN * 2;        // 16 MB
  bf16* Wcat  = (bf16*)p; p += (size_t)HIDDEN * QKV_N * 2;      // 24 MB
  bf16* Wob   = (bf16*)p; p += (size_t)HIDDEN * HIDDEN * 2;     // 8 MB
  bf16* QKV   = (bf16*)p; p += (size_t)TOK * QKV_N * 2;         // 48 MB
  bf16* Qr    = (bf16*)p; p += (size_t)TOK * NH  * HD * 2;      // 16 MB
  bf16* Kr    = (bf16*)p; p += (size_t)TOK * NKV * HD * 2;      // 8 MB
  bf16* AttnG = (bf16*)p; p += (size_t)TOK * NH  * HD * 2;      // 16 MB

  k_cvt_bf16 <<<(TOK * HIDDEN) / 256, 256, 0, stream>>>(x, Xb, TOK * HIDDEN);
  k_build_wcat<<<(HIDDEN * QKV_N) / 256, 256, 0, stream>>>(Wq, Wk, Wv, Wcat);
  k_cvt_bf16 <<<(HIDDEN * HIDDEN) / 256, 256, 0, stream>>>(Wo, Wob, HIDDEN * HIDDEN);

  // QKV projection: [4096 x 2048] x [2048 x 6144]
  k_gemm<bf16><<<dim3(QKV_N / 128, TOK / 128), 256, 0, stream>>>(Xb, Wcat, QKV, TOK, QKV_N, HIDDEN);

  // per-head RMSnorm + RoPE
  k_norm_rope<<<TOK, 256, 0, stream>>>(QKV, qw, kw, positions, Qr, Kr);

  // flash attention + gate
  k_attn<<<dim3(SEQQ / 128, NH, BATCH), 256, 0, stream>>>(Qr, Kr, QKV, amask, AttnG);

  // output projection: [4096 x 2048] x [2048 x 2048] -> fp32
  k_gemm<float><<<dim3(HIDDEN / 128, TOK / 128), 256, 0, stream>>>(AttnG, Wob, out, TOK, HIDDEN, HIDDEN);
}
